// Glove_Based_LSTM_Model_68204080660914
// MI455X (gfx1250) — matrix operand order repaired and checked by reading, not compiled, not run on hardware
//
#include <hip/hip_runtime.h>
#include <hip/hip_bf16.h>

#define VOCAB 400000
#define DIM   300
#define HID   300
#define NB    64      // batch
#define SEQ   512
#define GATES 1200    // 4*HID
#define KP    320     // K padded to multiple of 32 (h buffer stride)
#define KW    328     // LDS W stride (bank-conflict padding, 16B aligned)
#define NWG   19      // ceil(300/16) scan workgroups

typedef __attribute__((ext_vector_type(16))) __bf16 v16bf;
typedef __attribute__((ext_vector_type(8)))  __bf16 v8bf;
typedef __attribute__((ext_vector_type(8)))  float  v8f;
typedef __attribute__((ext_vector_type(4)))  float  v4f;
typedef __attribute__((ext_vector_type(4)))  int    v4i;

// Exact pointer types for the async global->LDS builtin (per compiler diag):
// param0: AS1 (global) v4i*, param1: AS3 (LDS) v4i*
typedef __attribute__((address_space(1))) v4i gv4i_t;
typedef __attribute__((address_space(3))) v4i lv4i_t;

#if __has_builtin(__builtin_amdgcn_global_load_async_to_lds_b128)
#define HAVE_ASYNC_LDS 1
#else
#define HAVE_ASYNC_LDS 0
#endif

__device__ __forceinline__ void wait_async0() {
#if __has_builtin(__builtin_amdgcn_s_wait_asynccnt)
    __builtin_amdgcn_s_wait_asynccnt(0);
#else
    asm volatile("s_wait_asynccnt 0" ::: "memory");
#endif
}

__device__ __forceinline__ float sigmoidf_(float x) { return 1.0f / (1.0f + __expf(-x)); }

// ---------------------------------------------------------------------------
// Kernel 1: fused embedding gather + x_proj GEMM (+ both biases)
//   x_proj[s*64+b][g] = sum_d glove[x[b][s]][d] * W_ih[g][d] + b_ih[g] + b_hh[g]
// One wave per 16x16 output tile; bf16 WMMA, f32 accumulate. HBM-bound
// (157MB x_proj write + ~39MB unique gathered rows, L2-resident re-reads).
// ---------------------------------------------------------------------------
__global__ __launch_bounds__(256) void xproj_wmma_kernel(
    const int* __restrict__ x, const float* __restrict__ glove,
    const float* __restrict__ W_ih, const float* __restrict__ b_ih,
    const float* __restrict__ b_hh, float* __restrict__ xproj)
{
    const int lane = threadIdx.x & 31;
    const int wave = threadIdx.x >> 5;
    const long tile = (long)blockIdx.x * 8 + wave;   // 153600 tiles total
    const int mtile = (int)(tile / 75);
    const int ntile = (int)(tile % 75);
    const int hi  = lane >> 4;
    const int l16 = lane & 15;

    // A-fragment source: gathered glove row for output row r = mtile*16 + (lane%16)
    const int row   = mtile * 16 + l16;              // r = s*64 + b
    const int s     = row >> 6;
    const int b     = row & 63;
    const int token = x[b * SEQ + s];
    const float* __restrict__ arow = glove + (long)token * DIM;

    // B-fragment source: W_ih row (gate column) g = ntile*16 + (lane%16)
    const int gcol = ntile * 16 + l16;
    const float* __restrict__ brow = W_ih + (long)gcol * DIM;

    v8f acc = {0.f, 0.f, 0.f, 0.f, 0.f, 0.f, 0.f, 0.f};

    for (int kc = 0; kc < 10; ++kc) {
        const int k0 = kc * 32;
        v16bf a, bm;
        if (kc < 9) {
            // fully in-range: contiguous vector loads (k = k0 + 16*hi + e in both A and B)
            const int ka = k0 + 16 * hi;
            v4f a0 = *(const v4f*)(arow + ka);
            v4f a1 = *(const v4f*)(arow + ka + 4);
            v4f a2 = *(const v4f*)(arow + ka + 8);
            v4f a3 = *(const v4f*)(arow + ka + 12);
            const int kb = k0 + 16 * hi;
            v4f b0 = *(const v4f*)(brow + kb);
            v4f b1 = *(const v4f*)(brow + kb + 4);
            v4f b2 = *(const v4f*)(brow + kb + 8);
            v4f b3 = *(const v4f*)(brow + kb + 12);
#pragma unroll
            for (int e = 0; e < 4; ++e) {
                a[e]       = (__bf16)a0[e];
                a[e + 4]   = (__bf16)a1[e];
                a[e + 8]   = (__bf16)a2[e];
                a[e + 12]  = (__bf16)a3[e];
                bm[e]      = (__bf16)b0[e];
                bm[e + 4]  = (__bf16)b1[e];
                bm[e + 8]  = (__bf16)b2[e];
                bm[e + 12] = (__bf16)b3[e];
            }
        } else {
            // K tail 288..319: clamp address (always in-row) + branchless select
#pragma unroll
            for (int e = 0; e < 8; ++e) {
                const int ka0 = k0 + 16 * hi + e;
                const int ka1 = ka0 + 8;
                const float va0 = arow[ka0 < DIM ? ka0 : DIM - 1];
                const float va1 = arow[ka1 < DIM ? ka1 : DIM - 1];
                a[e]     = (__bf16)(ka0 < DIM ? va0 : 0.0f);
                a[e + 8] = (__bf16)(ka1 < DIM ? va1 : 0.0f);
            }
#pragma unroll
            for (int e = 0; e < 16; ++e) {
                const int kb = k0 + 16 * hi + e;
                const float vb = brow[kb < DIM ? kb : DIM - 1];
                bm[e] = (__bf16)(kb < DIM ? vb : 0.0f);
            }
        }
        acc = __builtin_amdgcn_wmma_f32_16x16x32_bf16(
            false, a, false, bm, (short)0, acc, false, false);
    }

    const float bias = b_ih[gcol] + b_hh[gcol];
#pragma unroll
    for (int p = 0; p < 8; ++p) {                    // D: M = p + 8*hi, N = lane%16
        const long r = (long)mtile * 16 + p + 8 * hi;
        xproj[r * GATES + gcol] = acc[p] + bias;
    }
}

// ---------------------------------------------------------------------------
// Kernel 2: persistent LSTM scan. 19 workgroups, each owns 16 h-columns
// (last owns 12) and holds its W_hh slice (4 gates x 16 cols, bf16) in LDS.
// Per step: async-prefetch the step's x_proj slice into LDS (overlaps the
// WMMA chain), WMMA gates, fused cell update, h exchange through a global
// bf16 buffer, device-scope monotonic atomic barrier.
// ---------------------------------------------------------------------------
__global__ __launch_bounds__(256) void lstm_scan_kernel(
    const float* __restrict__ W_hh, const float* __restrict__ xproj,
    __bf16* hbuf, float* __restrict__ hfinal, unsigned int* barrier_cnt)
{
    __shared__ __attribute__((aligned(16))) __bf16 sW[64 * KW];  // [n][k], n=q*16+j (41KB)
    __shared__ float  sGate[4][NB][16];                          // raw gates (16KB)
    __shared__ float  sC[NB * 16];                               // cell state (4KB)
#if HAVE_ASYNC_LDS
    __shared__ __attribute__((aligned(16))) float sXp[4][NB][16]; // x_proj slice (16KB)
#endif

    const int t   = blockIdx.x;            // 0..18
    const int ct0 = t * 16;
    const int CT  = (HID - ct0 < 16) ? (HID - ct0) : 16;   // 16 or 12
    const int tid = threadIdx.x;

    // Stage W_hh slice into LDS as bf16, B-layout friendly ([n][k] contiguous in k)
    for (int idx = tid; idx < 64 * KP; idx += 256) {
        const int n = idx / KP, k = idx % KP;
        const int q = n >> 4, j = n & 15;
        float w = 0.0f;
        if (k < HID && (ct0 + j) < HID)
            w = W_hh[(long)(q * HID + ct0 + j) * HID + k];
        sW[n * KW + k] = (__bf16)w;
    }
    for (int idx = tid; idx < NB * 16; idx += 256) sC[idx] = 0.0f;
    __syncthreads();

    const int lane = tid & 31;
    const int wave = tid >> 5;
    const int hi   = lane >> 4;
    const int l16  = lane & 15;
    const int m    = wave & 3;             // batch tile 0..3 (rows m*16..m*16+15)
    const int q0   = (wave >> 2) * 2;      // this wave handles gates q0, q0+1

    const __bf16* hrow   = hbuf + (m * 16 + l16) * KP;     // A: M = batch row
    const __bf16* bbase0 = sW + (q0 * 16 + l16) * KW;      // B: N = gate col
    const __bf16* bbase1 = bbase0 + 16 * KW;

    for (int s = 0; s < SEQ; ++s) {
        const float* xp = xproj + (long)s * (NB * GATES);
#if HAVE_ASYNC_LDS
        // Issue async global->LDS copies of this step's x_proj slice; they
        // complete under ASYNCcnt while the WMMA chain below executes.
#pragma unroll
        for (int u = 0; u < 4; ++u) {
            const int seg = tid + 256 * u;        // 1024 x (4 floats)
            const int q   = seg >> 8;
            const int rem = seg & 255;
            const int bb  = rem >> 2;
            const int jd  = (rem & 3) * 4;
            if (ct0 + jd < HID) {
                const float* g = xp + (long)bb * GATES + q * HID + ct0 + jd;
                __builtin_amdgcn_global_load_async_to_lds_b128(
                    (gv4i_t*)(uintptr_t)g,
                    (lv4i_t*)(uintptr_t)&sXp[q][bb][jd], 0, 0);
                if (s + 1 < SEQ)
                    __builtin_prefetch(g + NB * GATES, 0, 1);  // warm L2 for next step
            }
        }
#endif
        v8f acc0 = {0.f,0.f,0.f,0.f,0.f,0.f,0.f,0.f};
        v8f acc1 = {0.f,0.f,0.f,0.f,0.f,0.f,0.f,0.f};
#pragma unroll
        for (int kc = 0; kc < 10; ++kc) {
            const int k0 = kc * 32;
            v16bf a, b0, b1;
            v8bf alo  = *(const v8bf*)(hrow   + k0 + 16 * hi);
            v8bf ahi  = *(const v8bf*)(hrow   + k0 + 16 * hi + 8);
            v8bf b0lo = *(const v8bf*)(bbase0 + k0 + 16 * hi);
            v8bf b0hi = *(const v8bf*)(bbase0 + k0 + 16 * hi + 8);
            v8bf b1lo = *(const v8bf*)(bbase1 + k0 + 16 * hi);
            v8bf b1hi = *(const v8bf*)(bbase1 + k0 + 16 * hi + 8);
#pragma unroll
            for (int e = 0; e < 8; ++e) {
                a[e] = alo[e];   a[e + 8] = ahi[e];
                b0[e] = b0lo[e]; b0[e + 8] = b0hi[e];
                b1[e] = b1lo[e]; b1[e + 8] = b1hi[e];
            }
            acc0 = __builtin_amdgcn_wmma_f32_16x16x32_bf16(
                false, a, false, b0, (short)0, acc0, false, false);
            acc1 = __builtin_amdgcn_wmma_f32_16x16x32_bf16(
                false, a, false, b1, (short)0, acc1, false, false);
        }
        // Scatter raw gates into LDS: D element p -> batch row m*16+p+8*hi, col lane%16
#pragma unroll
        for (int p = 0; p < 8; ++p) {
            const int bb = m * 16 + p + 8 * hi;
            sGate[q0][bb][l16]     = acc0[p];
            sGate[q0 + 1][bb][l16] = acc1[p];
        }
#if HAVE_ASYNC_LDS
        wait_async0();          // this wave's async LDS writes are done
#endif
        __syncthreads();

        // Fused cell update: 64x16 elements, 4 per thread
        for (int u = 0; u < 4; ++u) {
            const int idx = tid * 4 + u;
            const int bb = idx >> 4, j = idx & 15;
            if (j < CT) {
                const int col = ct0 + j;
#if HAVE_ASYNC_LDS
                const float xi = sXp[0][bb][j];
                const float xf = sXp[1][bb][j];
                const float xg = sXp[2][bb][j];
                const float xo = sXp[3][bb][j];
#else
                const long base = (long)bb * GATES + col;
                const float xi = xp[base];
                const float xf = xp[base + HID];
                const float xg = xp[base + 2 * HID];
                const float xo = xp[base + 3 * HID];
#endif
                const float gi = sigmoidf_(sGate[0][bb][j] + xi);
                const float gf = sigmoidf_(sGate[1][bb][j] + xf);
                const float gg = tanhf    (sGate[2][bb][j] + xg);
                const float go = sigmoidf_(sGate[3][bb][j] + xo);
                const float c  = gf * sC[idx] + gi * gg;
                sC[idx] = c;
                const float h = go * tanhf(c);
                hbuf[bb * KP + col] = (__bf16)h;
                if (s == SEQ - 1) hfinal[bb * HID + col] = h;
            }
        }
        __threadfence();       // make h stores device-visible before arriving
        __syncthreads();

        // Monotonic device-scope barrier across the 19 co-resident workgroups
        if (tid == 0) {
            atomicAdd(barrier_cnt, 1u);
            const unsigned target = (unsigned)(s + 1) * NWG;
            while (atomicAdd(barrier_cnt, 0u) < target) {
                __builtin_amdgcn_s_sleep(1);
            }
            __threadfence();   // acquire: invalidate before reading peers' h
        }
        __syncthreads();
    }
}

// ---------------------------------------------------------------------------
// Kernel 3: final FC  out[b][c] = h_last[b] . fc_w[c] + fc_b[c]   (64x3)
// ---------------------------------------------------------------------------
__global__ void fc_kernel(const float* __restrict__ hfinal,
                          const float* __restrict__ fc_w,
                          const float* __restrict__ fc_b,
                          float* __restrict__ out)
{
    const int idx = blockIdx.x * blockDim.x + threadIdx.x;
    if (idx >= NB * 3) return;
    const int b = idx / 3, c = idx % 3;
    float acc = fc_b[c];
    const float* hr = hfinal + b * HID;
    const float* wr = fc_w + c * HID;
    for (int k = 0; k < HID; ++k) acc = fmaf(hr[k], wr[k], acc);
    out[idx] = acc;
}

// Zero barrier counter + h exchange buffer each launch (determinism).
__global__ void init_kernel(unsigned int* cnt, __bf16* hbuf)
{
    const int tid = threadIdx.x;
    if (tid == 0) *cnt = 0u;
    for (int i = tid; i < NB * KP; i += 256) hbuf[i] = (__bf16)0.0f;
}

extern "C" void kernel_launch(void* const* d_in, const int* in_sizes, int n_in,
                              void* d_out, int out_size, void* d_ws, size_t ws_size,
                              hipStream_t stream)
{
    const int*   x     = (const int*)  d_in[0];
    const float* glove = (const float*)d_in[1];
    const float* W_ih  = (const float*)d_in[2];
    const float* W_hh  = (const float*)d_in[3];
    const float* b_ih  = (const float*)d_in[4];
    const float* b_hh  = (const float*)d_in[5];
    const float* fc_w  = (const float*)d_in[6];
    const float* fc_b  = (const float*)d_in[7];
    float* out = (float*)d_out;

    // Workspace layout
    char* ws = (char*)d_ws;
    unsigned int* cnt = (unsigned int*)ws;                        // [0, 256)
    __bf16* hbuf      = (__bf16*)(ws + 256);                      // 64*320 bf16 = 40960 B
    float*  hfinal    = (float*)(ws + 256 + 40960);               // 64*300 f32 = 76800 B
    float*  xproj     = (float*)(ws + 128 * 1024);                // 512*64*1200 f32 = 157 MB

    init_kernel<<<1, 256, 0, stream>>>(cnt, hbuf);
    // 2048 M-tiles * 75 N-tiles = 153600 waves, 8 waves/block
    xproj_wmma_kernel<<<19200, 256, 0, stream>>>(x, glove, W_ih, b_ih, b_hh, xproj);
    lstm_scan_kernel<<<NWG, 256, 0, stream>>>(W_hh, xproj, hbuf, hfinal, cnt);
    fc_kernel<<<1, 192, 0, stream>>>(hfinal, fc_w, fc_b, out);
}
